// MultiHeadAttention_5781025980740
// MI455X (gfx1250) — matrix operand order repaired and checked by reading, not compiled, not run on hardware
//
#include <hip/hip_runtime.h>

// Problem constants (match reference)
#define BB_ 4
#define S_  2048
#define DM_ 512
#define H_  8
#define DH_ 64

typedef __attribute__((ext_vector_type(16))) __bf16 v16bf;
typedef __attribute__((ext_vector_type(8)))  __bf16 v8bf;
typedef __attribute__((ext_vector_type(8)))  float  v8f;
typedef __attribute__((ext_vector_type(4)))  int    v4i;

// gfx1250 async global->LDS copy path (ASYNCcnt), guarded so the file still
// compiles (with the VGPR fallback) if the toolchain lacks the builtins.
#if defined(__HIP_DEVICE_COMPILE__) &&                                   \
    __has_builtin(__builtin_amdgcn_global_load_async_to_lds_b128) &&     \
    __has_builtin(__builtin_amdgcn_s_wait_asynccnt)
#define USE_ASYNC_COPY 1
typedef __attribute__((address_space(1))) v4i* g_b128_ptr;  // global (AS1)
typedef __attribute__((address_space(3))) v4i* l_b128_ptr;  // LDS    (AS3)
#else
#define USE_ASYNC_COPY 0
#endif

__device__ __forceinline__ v16bf frag_cat(v8bf lo, v8bf hi) {
  return __builtin_shufflevector(lo, hi, 0,1,2,3,4,5,6,7,8,9,10,11,12,13,14,15);
}
__device__ __forceinline__ v16bf frag_ld(const __bf16* p0, const __bf16* p1) {
  return frag_cat(*(const v8bf*)p0, *(const v8bf*)p1);
}
__device__ __forceinline__ v8f wmma_bf16(v16bf a, v16bf b, v8f c) {
  // D = A(16x32) * B(32x16) + C, f32 accumulate
  return __builtin_amdgcn_wmma_f32_16x16x32_bf16(false, a, false, b, (short)0, c, false, false);
}

// ---------------------------------------------------------------------------
// Tiled GEMM: C = A[M,K] * B[K,N] + bias, bf16 WMMA with f32 accumulate.
// Block = 256 threads (8 waves), tile 128(M) x 64(N), K-step 32.
// OUT_MODE 0: bf16, heads layout      out[((b*H+h)*S + s)*64 + d]     (q,k proj)
// OUT_MODE 2: f32,  flat [M, DM]      out[gm*512 + gn]                (final Y)
// OUT_MODE 3: bf16, heads TRANSPOSED  out[((b*H+h)*64 + d)*S + s]     (v proj)
// When A is already bf16 (OUT_MODE 2 over ctx), the A tile is staged with
// global_load_async_to_lds_b128 (pure copy, no conversion needed).
// ---------------------------------------------------------------------------
template<bool A_F32, int OUT_MODE>
__global__ __launch_bounds__(256) void gemm_wmma(
    const void* __restrict__ Ap, int lda,
    const float* __restrict__ Bp, int ldb,
    const float* __restrict__ bias,
    void* __restrict__ Op, int K)
{
  __shared__ __bf16 lsA[128 * 40];  // A tile 128x32, pitch 40 (80B rows, 16B aligned)
  __shared__ __bf16 lsB[64 * 40];   // B tile stored transposed: [n][k], pitch 40

  const float*  Af = (const float*)Ap;
  const __bf16* Ab = (const __bf16*)Ap;

  const int m0   = blockIdx.x * 128;
  const int n0   = blockIdx.y * 64;
  const int tid  = threadIdx.x;
  const int wave = tid >> 5;
  const int lane = tid & 31;
  const int lm   = lane & 15;
  const int lh   = lane >> 4;
  const int wm   = (wave >> 1) * 32;  // 0,32,64,96
  const int wn   = (wave & 1) * 32;   // 0,32

  v8f acc[2][2] = {};

  for (int k0 = 0; k0 < K; k0 += 32) {
    __syncthreads();
    // ---- Stage A tile ----
    if (A_F32) {
      // f32 -> bf16 conversion path (coalesced, 16 elements/thread)
      #pragma unroll
      for (int r = 0; r < 16; ++r) {
        int idx = tid + 256 * r;
        int row = idx >> 5, col = idx & 31;
        lsA[row * 40 + col] = (__bf16)Af[(size_t)(m0 + row) * lda + (k0 + col)];
      }
    } else {
#if USE_ASYNC_COPY
      // bf16 tile: async DMA global -> LDS, 16B chunks, 2 per thread
      #pragma unroll
      for (int r = 0; r < 2; ++r) {
        int idx = tid + 256 * r;           // 0..511
        int row = idx >> 2, ch = idx & 3;  // 128 rows x 4 x 16B
        const __bf16* gsrc = Ab + (size_t)(m0 + row) * lda + k0 + ch * 8;
        __builtin_amdgcn_global_load_async_to_lds_b128(
            (g_b128_ptr)gsrc,
            (l_b128_ptr)&lsA[row * 40 + ch * 8],
            0, 0);
      }
#else
      #pragma unroll
      for (int r = 0; r < 16; ++r) {
        int idx = tid + 256 * r;
        int row = idx >> 5, col = idx & 31;
        lsA[row * 40 + col] = Ab[(size_t)(m0 + row) * lda + (k0 + col)];
      }
#endif
    }
    // ---- Stage B tile transposed (weights are f32 [K,N] row-major) ----
    #pragma unroll
    for (int r = 0; r < 8; ++r) {
      int idx = tid + 256 * r;
      int krow = idx >> 6, col = idx & 63;
      float bv = Bp[(size_t)(k0 + krow) * ldb + (n0 + col)];
      lsB[col * 40 + krow] = (__bf16)bv;
    }
#if USE_ASYNC_COPY
    if (!A_F32) __builtin_amdgcn_s_wait_asynccnt(0);
#endif
    __syncthreads();

    v16bf afr[2], bfr[2];
    #pragma unroll
    for (int mi = 0; mi < 2; ++mi) {
      const __bf16* pa = &lsA[(wm + mi * 16 + lm) * 40];
      afr[mi] = frag_ld(pa + lh * 8, pa + 16 + lh * 8);
    }
    #pragma unroll
    for (int ni = 0; ni < 2; ++ni) {
      const __bf16* pb = &lsB[(wn + ni * 16 + lm) * 40 + lh * 8];
      bfr[ni] = frag_ld(pb, pb + 16);
    }
    #pragma unroll
    for (int mi = 0; mi < 2; ++mi)
      #pragma unroll
      for (int ni = 0; ni < 2; ++ni)
        acc[mi][ni] = wmma_bf16(afr[mi], bfr[ni], acc[mi][ni]);
  }

  // Epilogue: C/D layout -> lane (lm,lh), VGPR r maps to M = r + 8*lh, N = lm
  #pragma unroll
  for (int mi = 0; mi < 2; ++mi) {
    #pragma unroll
    for (int ni = 0; ni < 2; ++ni) {
      int gn = n0 + wn + ni * 16 + lm;
      float bv = bias ? bias[gn] : 0.0f;
      #pragma unroll
      for (int r = 0; r < 8; ++r) {
        int gm = m0 + wm + mi * 16 + 8 * lh + r;
        float val = acc[mi][ni][r] + bv;
        if (OUT_MODE == 0) {
          int bb = gm >> 11, s = gm & (S_ - 1);
          int h = gn >> 6, d = gn & 63;
          ((__bf16*)Op)[(((size_t)bb * H_ + h) * S_ + s) * DH_ + d] = (__bf16)val;
        } else if (OUT_MODE == 3) {
          int bb = gm >> 11, s = gm & (S_ - 1);
          int h = gn >> 6, d = gn & 63;
          ((__bf16*)Op)[(((size_t)bb * H_ + h) * DH_ + d) * S_ + s] = (__bf16)val;
        } else {
          ((float*)Op)[(size_t)gm * DM_ + gn] = val;
        }
      }
    }
  }
}

// ---------------------------------------------------------------------------
// Fused attention: per block (256 thr, 8 waves) handles 16 query rows of one
// (b,h):  phase1 scores 16x2048 -> LDS (WMMA),  phase2 exact softmax in LDS
// (+ single nontemporal write of probs to d_out),  phase3 ctx = P @ V with
// K-split across waves and ds_add_f32 reduction.
// ---------------------------------------------------------------------------
__global__ __launch_bounds__(256) void attn_fused(
    const void* __restrict__ qpv, const void* __restrict__ kpv,
    const void* __restrict__ vtv, const float* __restrict__ mask,
    float* __restrict__ att, void* __restrict__ ctxv)
{
  constexpr int PITCH = 2056;          // 2048 + 8 f32 pad (32B aligned rows)
  __shared__ float sc[16 * PITCH];     // ~131.6 KB score/prob stripe
  __shared__ float red[16 * DH_];      // 4 KB cross-wave ctx reduction

  const __bf16* qp = (const __bf16*)qpv;
  const __bf16* kp = (const __bf16*)kpv;
  const __bf16* vt = (const __bf16*)vtv;
  __bf16* ctx = (__bf16*)ctxv;

  const int m0  = blockIdx.x * 16;
  const int z   = blockIdx.y;          // b*H + h
  const int bb  = z >> 3, hh = z & 7;
  const int tid = threadIdx.x;
  const int wave = tid >> 5, lane = tid & 31;
  const int lm = lane & 15, lh = lane >> 4;

  const __bf16* qh = qp + (size_t)z * S_ * DH_;   // [s][d]
  const __bf16* kh = kp + (size_t)z * S_ * DH_;   // [s][d]
  const __bf16* vh = vt + (size_t)z * DH_ * S_;   // [d][s] (transposed)

  // Warm L2 with this wave's phase-3 V panel (64 rows x 512B K-slice) while
  // phases 1/2 run. V is 8MB total vs 192MB L2 -> stays resident.
  {
    const char* vbase = (const char*)vh;
    #pragma unroll
    for (int i = 0; i < 8; ++i) {
      int row = ((lane & 7) << 3) + i;                       // 0..63
      const char* p = vbase + ((size_t)row * S_ + wave * 256) * 2
                            + ((size_t)(lane >> 3) << 7);    // 4 x 128B lines
      __builtin_prefetch(p, 0, 1);
    }
  }

  // ---- Phase 1: raw scores = q . k^T for 16 rows x 2048 cols ----
  const __bf16* qrow = qh + (size_t)(m0 + lm) * DH_;
  v16bf aq0 = frag_ld(qrow + lh * 8,      qrow + 16 + lh * 8);       // K=0..31
  v16bf aq1 = frag_ld(qrow + 32 + lh * 8, qrow + 48 + lh * 8);       // K=32..63

  #pragma unroll
  for (int ni = 0; ni < 16; ++ni) {
    const int n = wave * 256 + ni * 16 + lm;
    const __bf16* krow = kh + (size_t)n * DH_;
    v16bf b0 = frag_ld(krow + lh * 8,      krow + 16 + lh * 8);
    v16bf b1 = frag_ld(krow + 32 + lh * 8, krow + 48 + lh * 8);
    v8f acc = {};
    acc = wmma_bf16(aq0, b0, acc);
    acc = wmma_bf16(aq1, b1, acc);
    #pragma unroll
    for (int r = 0; r < 8; ++r)
      sc[(r + 8 * lh) * PITCH + n] = acc[r];
  }
  // zero the ctx reduction buffer while we're at it
  #pragma unroll
  for (int i = tid; i < 16 * DH_; i += 256) red[i] = 0.0f;
  __syncthreads();

  // ---- Phase 2: exact softmax; 2 rows per wave; write probs to HBM once ----
  const float scale = 0.125f;  // 1/sqrt(64)
  #pragma unroll
  for (int rr = 0; rr < 2; ++rr) {
    const int row = wave * 2 + rr;
    float x[64];
    #pragma unroll
    for (int j = 0; j < 64; ++j) {
      int col = lane + 32 * j;
      x[j] = sc[row * PITCH + col] * scale + mask[(size_t)bb * S_ + col] * -1e9f;
    }
    float mx = x[0];
    #pragma unroll
    for (int j = 1; j < 64; ++j) mx = fmaxf(mx, x[j]);
    #pragma unroll
    for (int off = 16; off > 0; off >>= 1) mx = fmaxf(mx, __shfl_xor(mx, off, 32));
    float s = 0.0f;
    #pragma unroll
    for (int j = 0; j < 64; ++j) { x[j] = __expf(x[j] - mx); s += x[j]; }
    #pragma unroll
    for (int off = 16; off > 0; off >>= 1) s += __shfl_xor(s, off, 32);
    const float inv = 1.0f / s;
    float* gout = att + ((size_t)z * S_ + m0 + row) * S_;
    #pragma unroll
    for (int j = 0; j < 64; ++j) {
      int col = lane + 32 * j;
      float p = x[j] * inv;
      sc[row * PITCH + col] = p;                  // keep probs for phase 3
      __builtin_nontemporal_store(p, &gout[col]); // write-once stream -> NT
    }
  }
  __syncthreads();

  // ---- Phase 3: ctx(16x64) = P(16x2048) @ V(2048x64); K split across waves ----
  v8f cacc[4] = {};
  const int kbase = wave * 256;
  #pragma unroll
  for (int ks = 0; ks < 8; ++ks) {
    const int k0 = kbase + ks * 32;
    // A fragment: probs f32 -> bf16 on the fly (contiguous 32B runs)
    v8f plo = *(const v8f*)&sc[lm * PITCH + k0 + lh * 8];
    v8f phi = *(const v8f*)&sc[lm * PITCH + k0 + 16 + lh * 8];
    v16bf a;
    #pragma unroll
    for (int i = 0; i < 8; ++i) { a[i] = (__bf16)plo[i]; a[i + 8] = (__bf16)phi[i]; }
    #pragma unroll
    for (int ni = 0; ni < 4; ++ni) {
      const __bf16* pv = vh + (size_t)(ni * 16 + lm) * S_ + k0 + lh * 8;
      v16bf b = frag_ld(pv, pv + 16);
      cacc[ni] = wmma_bf16(a, b, cacc[ni]);
    }
  }
  #pragma unroll
  for (int ni = 0; ni < 4; ++ni)
    #pragma unroll
    for (int r = 0; r < 8; ++r)
      atomicAdd(&red[(r + 8 * lh) * DH_ + ni * 16 + lm], cacc[ni][r]);  // ds_add_f32
  __syncthreads();

  // write ctx tile [16 x 64] as bf16, flat [B*S, DM] layout for the Wo GEMM
  #pragma unroll
  for (int i = tid; i < 16 * DH_; i += 256) {
    int row = i >> 6, col = i & 63;
    ctx[((size_t)bb * S_ + m0 + row) * DM_ + hh * DH_ + col] = (__bf16)red[i];
  }
}

// ---------------------------------------------------------------------------
extern "C" void kernel_launch(void* const* d_in, const int* in_sizes, int n_in,
                              void* d_out, int out_size, void* d_ws, size_t ws_size,
                              hipStream_t stream) {
  (void)in_sizes; (void)n_in; (void)out_size; (void)ws_size;
  const float* Q    = (const float*)d_in[0];
  const float* Kin  = (const float*)d_in[1];
  const float* V    = (const float*)d_in[2];
  const float* mask = (const float*)d_in[3];
  const float* Wq   = (const float*)d_in[4];
  const float* bq   = (const float*)d_in[5];
  const float* Wk   = (const float*)d_in[6];
  const float* bk   = (const float*)d_in[7];
  const float* Wv   = (const float*)d_in[8];
  const float* bv   = (const float*)d_in[9];
  const float* Wo   = (const float*)d_in[10];
  const float* bo   = (const float*)d_in[11];

  float* Y   = (float*)d_out;                               // [4,2048,512]
  float* att = Y + (size_t)BB_ * S_ * DM_;                  // [4,8,2048,2048]

  // workspace: 4 x 8MB bf16 buffers
  char* ws   = (char*)d_ws;
  void* qp   = (void*)(ws);                                 // [B,H,S,64] bf16
  void* kp   = (void*)(ws + (size_t)8  * 1024 * 1024);      // [B,H,S,64] bf16
  void* vT   = (void*)(ws + (size_t)16 * 1024 * 1024);      // [B,H,64,S] bf16
  void* ctx  = (void*)(ws + (size_t)24 * 1024 * 1024);      // [B*S,512]  bf16

  dim3 thr(256);
  dim3 gProj((BB_ * S_) / 128, DM_ / 64, 1);                // (64, 8)

  gemm_wmma<true, 0><<<gProj, thr, 0, stream>>>(Q,   DM_, Wq, DM_, bq, qp, DM_);
  gemm_wmma<true, 0><<<gProj, thr, 0, stream>>>(Kin, DM_, Wk, DM_, bk, kp, DM_);
  gemm_wmma<true, 3><<<gProj, thr, 0, stream>>>(V,   DM_, Wv, DM_, bv, vT, DM_);

  dim3 gAtt(S_ / 16, BB_ * H_, 1);                          // (128, 32)
  attn_fused<<<gAtt, thr, 0, stream>>>(qp, kp, vT, mask, att, ctx);

  gemm_wmma<false, 2><<<gProj, thr, 0, stream>>>(ctx, DM_, Wo, DM_, bo, Y, DM_);
}
